// MultiHeadSelfAttention_51719996178850
// MI455X (gfx1250) — compile-verified, matrix operand order checked by reading, not run on hardware
//
#include <hip/hip_runtime.h>
#include <hip/hip_bf16.h>
#include <math.h>

// Problem constants (from reference)
#define B_  4
#define S_  2048
#define D_  2048
#define H_  16
#define DK_ 128

typedef __attribute__((ext_vector_type(16))) _Float16     v16h;
typedef __attribute__((ext_vector_type(8)))  float        v8f;
typedef __attribute__((ext_vector_type(4)))  unsigned int v4u;
typedef __attribute__((ext_vector_type(4)))  float        v4f;
typedef __attribute__((ext_vector_type(4)))  _Float16     v4h;

// ---------------------------------------------------------------------------
// CDNA5 WMMA helpers (wave32, 16x16x32 f16 -> f32)
// ---------------------------------------------------------------------------
static __device__ __forceinline__ v8f wmma_f16(v16h a, v16h b, v8f c) {
  // (neg_a, A, neg_b, B, c_mod, C, reuse_a, reuse_b)
  return __builtin_amdgcn_wmma_f32_16x16x32_f16(false, a, false, b, (short)0, c,
                                                false, false);
}

// Load one A/B fragment for 16x16x32 f16 WMMA.
// A layout: lane holds row m = lane&15; K halves: lanes0-15 -> K 0-7,16-23,
// lanes16-31 -> K 8-15,24-31.  B layout is symmetric with n = lane&15.
// `row` = pointer to element [row m (or col n)][k=0] of the 16x32 fragment;
// the 32 K-values must be contiguous. Requires 16B alignment of `row`.
static __device__ __forceinline__ v16h load_frag(const _Float16* row, int lane) {
  const int off = (lane & 16) ? 8 : 0;
  union { v4u q[2]; v16h h; } u;
  u.q[0] = *(const v4u*)(row + off);        // K-halves 0..7 of this lane
  u.q[1] = *(const v4u*)(row + 16 + off);   // K-halves 8..15 of this lane
  return u.h;
}

// Async global->LDS copy of 32 bytes per lane (CDNA5 GLOBAL_LOAD_ASYNC_TO_LDS,
// tracked by ASYNCcnt; bypasses VGPRs). The instruction offset applies to both
// the LDS and the global address, so offset:0/16 covers the 32B chunk.
// `ldsoff` = LDS byte offset (low 32 bits of the flat shared-pointer address).
static __device__ __forceinline__ void async_copy32(unsigned ldsoff,
                                                    const void* gsrc) {
  asm volatile(
      "global_load_async_to_lds_b128 %0, %1, off\n\t"
      "global_load_async_to_lds_b128 %0, %1, off offset:16"
      :
      : "v"(ldsoff), "v"(gsrc)
      : "memory");
}

static __device__ __forceinline__ void async_wait0() {
  asm volatile("s_wait_asynccnt 0" ::: "memory");
}

// Flat shared-pointer -> LDS byte offset (aperture lives in the high 32 bits).
static __device__ __forceinline__ unsigned lds_off(const void* p) {
  return (unsigned)(unsigned long long)p;
}

// ---------------------------------------------------------------------------
// fp32 -> fp16 conversion (vectorized x4)
// ---------------------------------------------------------------------------
__global__ void cvt_f32_f16(const v4f* __restrict__ in, v4h* __restrict__ out,
                            size_t n4) {
  size_t i = (size_t)blockIdx.x * blockDim.x + threadIdx.x;
  if (i < n4) out[i] = __builtin_convertvector(in[i], v4h);
}

// ---------------------------------------------------------------------------
// C[m,n] = sum_k A[m,k] * W[n,k]   (A:[M,K] f16, W:[N,K] f16)
// 256 threads = 8 waves; 128x128 tile; wave w owns rows 16w..16w+15.
// W tile double-buffered in LDS; async DMA of tile i+1 overlaps WMMA of tile i.
// ---------------------------------------------------------------------------
template <bool OUT_HALF>
__global__ __launch_bounds__(256) void gemm_xWT(const _Float16* __restrict__ A,
                                                const _Float16* __restrict__ W,
                                                void* __restrict__ C, int M,
                                                int N, int K) {
  __shared__ _Float16 bt[2][128][40];  // W tile, [n][k], padded 80B stride
  const int tid = threadIdx.x;
  const int lane = tid & 31;
  const int w = tid >> 5;
  const int n0 = blockIdx.x * 128;
  const int m0 = blockIdx.y * 128;
  const int arow = m0 + w * 16 + (lane & 15);

  v8f acc[8] = {};

  const int ln = tid & 127;          // cooperative-loader row
  const int lk = (tid >> 7) * 16;    // cooperative-loader k-part (0 or 16)
  const unsigned bt_off[2] = {lds_off(&bt[0][ln][lk]), lds_off(&bt[1][ln][lk])};
  const _Float16* Wl = W + (size_t)(n0 + ln) * K + lk;  // this thread's W row

  const int nk = K / 32;
  async_copy32(bt_off[0], Wl);  // prologue: tile 0 in flight

  for (int i = 0; i < nk; ++i) {
    const int k0 = i * 32;
    const int cur = i & 1;
    async_wait0();        // tile i landed in LDS (this wave's share)
    __syncthreads();      // all waves: tile i visible; prev reads of buf done
    if (i + 1 < nk) async_copy32(bt_off[cur ^ 1], Wl + k0 + 32);

    // prefetch all fragments, then fire the WMMA chain
    v16h af = load_frag(A + (size_t)arow * K + k0, lane);
    v16h bf[8];
    for (int t = 0; t < 8; ++t)
      bf[t] = load_frag(&bt[cur][t * 16 + (lane & 15)][0], lane);
    for (int t = 0; t < 8; ++t) acc[t] = wmma_f16(af, bf[t], acc[t]);
    __syncthreads();      // reads of bt[cur] done before it is overwritten
  }

  // Epilogue: C layout -> lane: n = lane&15, VGPR j: m = j + (lane[4]?8:0)
  const int mhi = (lane & 16) ? 8 : 0;
  const size_t mbase = (size_t)(m0 + w * 16 + mhi);
  for (int t = 0; t < 8; ++t) {
    const int ncol = n0 + t * 16 + (lane & 15);
    for (int j = 0; j < 8; ++j) {
      size_t idx = (mbase + j) * (size_t)N + ncol;
      if (OUT_HALF)
        ((_Float16*)C)[idx] = (_Float16)acc[t][j];
      else
        ((float*)C)[idx] = acc[t][j];
    }
  }
}

// ---------------------------------------------------------------------------
// RoPE applied in place to f16 Q and K in [B, S, H*DK] layout.
// One thread per (b, s, h, pair).
// ---------------------------------------------------------------------------
__global__ void rope_qk(_Float16* __restrict__ q, _Float16* __restrict__ k,
                        const int* __restrict__ pos, size_t total) {
  size_t i = (size_t)blockIdx.x * blockDim.x + threadIdx.x;
  if (i >= total) return;
  const int p = (int)(i & 63);            // pair index 0..63
  const int h = (int)((i >> 6) & (H_ - 1));
  const int s = (int)((i >> 10) & (S_ - 1));
  const int b = (int)(i >> 21);
  const float fp = (float)pos[s];
  // theta^(-2p/DK) = exp(p * (-ln(10000)/64))
  const float ang = fp * __expf((float)p * (-0.14391157f));
  const float cs = __cosf(ang), sn = __sinf(ang);
  const size_t base =
      ((size_t)(b * S_ + s)) * (size_t)D_ + (size_t)h * DK_ + 2 * p;
  {
    float x0 = (float)q[base], x1 = (float)q[base + 1];
    q[base]     = (_Float16)(cs * x0 - sn * x1);
    q[base + 1] = (_Float16)(sn * x0 + cs * x1);
  }
  {
    float x0 = (float)k[base], x1 = (float)k[base + 1];
    k[base]     = (_Float16)(cs * x0 - sn * x1);
    k[base + 1] = (_Float16)(sn * x0 + cs * x1);
  }
}

// ---------------------------------------------------------------------------
// V [B, S, H*DK] -> Vt [B*H, DK, S]  (so PV B-fragments are row-contiguous)
// ---------------------------------------------------------------------------
__global__ void vtrans(const _Float16* __restrict__ v,
                       _Float16* __restrict__ vt, size_t total) {
  size_t i = (size_t)blockIdx.x * blockDim.x + threadIdx.x;
  if (i >= total) return;
  const int c = (int)(i & (D_ - 1));
  const int s = (int)((i >> 11) & (S_ - 1));
  const int b = (int)(i >> 22);
  const int h = c >> 7, d = c & (DK_ - 1);
  vt[(((size_t)(b * H_ + h) * DK_ + d) << 11) + s] = v[i];
}

// ---------------------------------------------------------------------------
// Flash attention, causal. Block = (b*H+h, 128-query tile), 8 waves x 16 rows.
// Q,K: [B,S,H*DK] f16 (RoPE'd).  Vt: [B*H, DK, S] f16.  O: [B,S,H*DK] f16.
// K/V tiles double-buffered in LDS; async DMA of block i+1 overlaps the
// score/softmax/PV compute of block i.  sc/ph/al are per-wave regions: LDS ops
// of one wave complete in order, so no block barrier is needed around softmax.
// ---------------------------------------------------------------------------
__global__ __launch_bounds__(256) void attn(const _Float16* __restrict__ Q,
                                            const _Float16* __restrict__ Kk,
                                            const _Float16* __restrict__ Vt,
                                            _Float16* __restrict__ O) {
  __shared__ _Float16 kt[2][32][136];   // K tile [key][dk], 272B stride
  __shared__ _Float16 vs[2][128][40];   // V tile [dk][key], 80B stride
  __shared__ float    sc[8][16][32];    // score tile per wave
  __shared__ _Float16 ph[8][16][40];    // probs (padded, 16B-aligned rows)
  __shared__ float    al[8][16];        // per-row alpha (reused as 1/l at end)

  const int tid = threadIdx.x, lane = tid & 31, w = tid >> 5;
  const int bh = blockIdx.x;
  const int b = bh >> 4, h = bh & (H_ - 1);
  const int qb = blockIdx.y;
  const int q0 = qb * 128 + w * 16;      // wave's first query row
  const float scale = 0.08838834764831845f;  // 1/sqrt(128)

  // Load Q fragments once: 16 queries x DK=128 -> 4 fragments of 16x32
  const _Float16* Qb =
      Q + ((size_t)(b * S_ + q0 + (lane & 15))) * (size_t)D_ + h * DK_;
  v16h qf[4];
  for (int c = 0; c < 4; ++c) qf[c] = load_frag(Qb + c * 32, lane);

  v8f acc[8] = {};
  float mrun = -3.0e38f, lrun = 0.0f;  // row stats (valid in lanes 0..15)
  const int mhi = (lane & 16) ? 8 : 0;
  const int nkb = (qb + 1) * 4;        // uniform across the block -> syncs legal

  // cooperative staging indices
  const int kr = tid >> 3, kp = (tid & 7) * 16;   // K tile: 32 x 128 halves
  const int vr = tid >> 1, vp = (tid & 1) * 16;   // V tile: 128 x 32 halves
  const unsigned kt_off[2] = {lds_off(&kt[0][kr][kp]), lds_off(&kt[1][kr][kp])};
  const unsigned vs_off[2] = {lds_off(&vs[0][vr][vp]), lds_off(&vs[1][vr][vp])};
  const _Float16* kg0 =
      Kk + ((size_t)(b * S_ + kr)) * (size_t)D_ + h * DK_ + kp;
  const _Float16* vg0 = Vt + ((size_t)bh * DK_ + vr) * (size_t)S_ + vp;

  // prologue: key block 0 in flight
  async_copy32(kt_off[0], kg0);
  async_copy32(vs_off[0], vg0);

  for (int kb = 0; kb < nkb; ++kb) {
    const int k0 = kb * 32;
    const int cur = kb & 1;

    async_wait0();        // block kb landed (this wave's share)
    __syncthreads();      // all waves: tiles visible; prev buf reads done
    if (kb + 1 < nkb) {   // overlap DMA of block kb+1 with compute of kb
      async_copy32(kt_off[cur ^ 1], kg0 + (size_t)(k0 + 32) * D_);
      async_copy32(vs_off[cur ^ 1], vg0 + (k0 + 32));
    }

    // --- scores S = Q K^T : prefetch 8 B-fragments, then 8 WMMAs ----------
    v16h kf[8];
    for (int ns = 0; ns < 2; ++ns)
      for (int cc = 0; cc < 4; ++cc)
        kf[ns * 4 + cc] =
            load_frag(&kt[cur][ns * 16 + (lane & 15)][cc * 32], lane);
    v8f c0 = {}, c1 = {};
    for (int cc = 0; cc < 4; ++cc) {
      c0 = wmma_f16(qf[cc], kf[cc], c0);
      c1 = wmma_f16(qf[cc], kf[4 + cc], c1);
    }
    for (int ns = 0; ns < 2; ++ns) {
      const v8f cv = ns ? c1 : c0;
      const int n = ns * 16 + (lane & 15);
      const int key = k0 + n;
      for (int j = 0; j < 8; ++j) {
        const int qrow = q0 + mhi + j;
        sc[w][mhi + j][n] = (key <= qrow) ? cv[j] * scale : -3.0e38f;
      }
    }

    // --- online softmax (lanes 0..15 own one row each; per-wave LDS) ------
    if (lane < 16) {
      const int r = lane;
      float bm = -3.0e38f;
      for (int n = 0; n < 32; ++n) bm = fmaxf(bm, sc[w][r][n]);
      const float mnew = fmaxf(mrun, bm);
      const float a = __expf(mrun - mnew);
      const int qrow = q0 + r;
      float ssum = 0.0f;
      for (int n = 0; n < 32; ++n) {
        float p = (k0 + n <= qrow) ? __expf(sc[w][r][n] - mnew) : 0.0f;
        ph[w][r][n] = (_Float16)p;
        ssum += p;
      }
      lrun = lrun * a + ssum;
      mrun = mnew;
      al[w][r] = a;
    }

    // --- rescale running O, then O += P V ---------------------------------
    {
      float aj[8];
      for (int j = 0; j < 8; ++j) aj[j] = al[w][mhi + j];
      for (int t = 0; t < 8; ++t)
        for (int j = 0; j < 8; ++j) acc[t][j] *= aj[j];
    }
    v16h pf = load_frag(&ph[w][lane & 15][0], lane);
    v16h vf[8];
    for (int t = 0; t < 8; ++t)
      vf[t] = load_frag(&vs[cur][t * 16 + (lane & 15)][0], lane);
    for (int t = 0; t < 8; ++t) acc[t] = wmma_f16(pf, vf[t], acc[t]);

    __syncthreads();      // reads of buf[cur] done before it is overwritten
  }

  // --- finalize: O /= l  (al reused as 1/l; per-wave, no barrier needed) ---
  if (lane < 16) al[w][lane] = (lrun > 0.0f) ? 1.0f / lrun : 0.0f;
  float lj[8];
  for (int j = 0; j < 8; ++j) lj[j] = al[w][mhi + j];
  _Float16* Ob =
      O + ((size_t)(b * S_ + q0 + mhi)) * (size_t)D_ + h * DK_ + (lane & 15);
  for (int t = 0; t < 8; ++t)
    for (int j = 0; j < 8; ++j)
      Ob[(size_t)j * D_ + t * 16] = (_Float16)(acc[t][j] * lj[j]);
}

// ---------------------------------------------------------------------------
// Host-side orchestration
// ---------------------------------------------------------------------------
extern "C" void kernel_launch(void* const* d_in, const int* in_sizes, int n_in,
                              void* d_out, int out_size, void* d_ws,
                              size_t ws_size, hipStream_t stream) {
  (void)in_sizes; (void)n_in; (void)out_size; (void)ws_size;
  const float* x  = (const float*)d_in[0];
  const float* Wq = (const float*)d_in[1];
  const float* Wk = (const float*)d_in[2];
  const float* Wv = (const float*)d_in[3];
  const float* Wo = (const float*)d_in[4];
  const int*  pos = (const int*)d_in[5];

  char* ws = (char*)d_ws;
  const size_t MB = 1ull << 20;
  _Float16* x_h  = (_Float16*)(ws);             // 32 MB : x fp16
  _Float16* wq_h = (_Float16*)(ws + 32 * MB);   //  8 MB
  _Float16* wk_h = (_Float16*)(ws + 40 * MB);   //  8 MB
  _Float16* wv_h = (_Float16*)(ws + 48 * MB);   //  8 MB
  _Float16* wo_h = (_Float16*)(ws + 56 * MB);   //  8 MB
  _Float16* q_h  = (_Float16*)(ws + 64 * MB);   // 32 MB : Q [B,S,D]
  _Float16* k_h  = (_Float16*)(ws + 96 * MB);   // 32 MB : K [B,S,D]
  _Float16* v_h  = (_Float16*)(ws + 128 * MB);  // 32 MB : V [B,S,D]
  _Float16* vt_h = (_Float16*)(ws + 160 * MB);  // 32 MB : Vt [B*H,DK,S]
  _Float16* ao_h = v_h;  // attention output reuses V (V dead after vtrans)

  const size_t M  = (size_t)B_ * S_;            // 8192
  const size_t NX = M * D_;                     // x elements
  const size_t NW = (size_t)D_ * D_;            // weight elements

  // fp32 -> fp16
  cvt_f32_f16<<<(unsigned)(NX / 4 / 256), 256, 0, stream>>>((const v4f*)x,
                                                            (v4h*)x_h, NX / 4);
  cvt_f32_f16<<<(unsigned)(NW / 4 / 256), 256, 0, stream>>>((const v4f*)Wq,
                                                            (v4h*)wq_h, NW / 4);
  cvt_f32_f16<<<(unsigned)(NW / 4 / 256), 256, 0, stream>>>((const v4f*)Wk,
                                                            (v4h*)wk_h, NW / 4);
  cvt_f32_f16<<<(unsigned)(NW / 4 / 256), 256, 0, stream>>>((const v4f*)Wv,
                                                            (v4h*)wv_h, NW / 4);
  cvt_f32_f16<<<(unsigned)(NW / 4 / 256), 256, 0, stream>>>((const v4f*)Wo,
                                                            (v4h*)wo_h, NW / 4);

  // Q/K/V projections: [8192,2048] x [2048,2048]^T, f16 out
  dim3 ggrid(D_ / 128, (unsigned)(M / 128));
  gemm_xWT<true><<<ggrid, 256, 0, stream>>>(x_h, wq_h, q_h, (int)M, D_, D_);
  gemm_xWT<true><<<ggrid, 256, 0, stream>>>(x_h, wk_h, k_h, (int)M, D_, D_);
  gemm_xWT<true><<<ggrid, 256, 0, stream>>>(x_h, wv_h, v_h, (int)M, D_, D_);

  // RoPE on Q and K (in place)
  const size_t nrope = (size_t)B_ * S_ * H_ * (DK_ / 2);  // 8,388,608
  rope_qk<<<(unsigned)(nrope / 256), 256, 0, stream>>>(q_h, k_h, pos, nrope);

  // V transpose to [B*H, DK, S]
  vtrans<<<(unsigned)(NX / 256), 256, 0, stream>>>(v_h, vt_h, NX);

  // Causal flash attention
  dim3 agrid(B_ * H_, S_ / 128);
  attn<<<agrid, 256, 0, stream>>>(q_h, k_h, vt_h, ao_h);

  // Output projection -> fp32 d_out
  gemm_xWT<false><<<ggrid, 256, 0, stream>>>(ao_h, wo_h, (float*)d_out, (int)M,
                                             D_, D_);
}
